// GRU_29815662969384
// MI455X (gfx1250) — matrix operand order repaired and checked by reading, not compiled, not run on hardware
//
#include <hip/hip_runtime.h>
#include <hip/hip_bf16.h>

// ---------------------------------------------------------------------------
// GRU scan for MI455X (gfx1250), wave32 + v_wmma_f32_16x16x32_bf16.
//
// Layouts (packed by prep kernels into d_ws):
//   Bh  : hidden weights, B-operand layout, tiles [kt(16)][nt(96)], each tile
//         = 32 lanes * 32 bytes (16 bf16/lane: lane = n + 16*((k>>3)&1),
//         vgpr v holds k = 8h+2v, 8h+2v+1, +8 for v>=4, h = lane/16)
//   Bi  : input weights, same, tiles [kt(8)][nt(96)]
//   bias: [ b_ir+b_hr | b_iz+b_hz | b_in | b_hn ]  (4 * 512 f32)
//   XE  : gathered embeddings in A-operand layout, tiles
//         [t(1024)][wg(4)][kt(8)], each tile 32 lanes * 32B
//         (lane = m + 16*((ko>>3)&1), chunk c = ko>>4, pos p = ko&7)
// Gate tile streams: for a fixed kt, the R/Z/N tiles of one h-column block
// sit at +0 / +32768 / +65536 bytes (32 column-tiles apart), and consecutive
// kt rows are +98304 bytes -> one running pointer + immediate offsets.
// ---------------------------------------------------------------------------

typedef __attribute__((ext_vector_type(16))) __bf16 v16bf;
typedef __attribute__((ext_vector_type(8)))  float  v8f;

#define E_DIM 256
#define H_DIM 512
#define B_DIM 64
#define T_DIM 1024
#define NG    1536          // 3*H gate columns
#define NT    96            // NG/16 column tiles
#define KT_H  16            // H/32 K-tiles (hidden)
#define KT_E  8             // E/32 K-tiles (input)

#define ZOFF   32768        // byte distance R->Z gate tile (32 tiles * 1024B)
#define NOFF   65536        // byte distance R->N gate tile
#define KSTRIDE 98304       // byte distance between kt rows (96 tiles * 1024B)

// workspace offsets (bytes)
#define OFF_BH   0u
#define SZ_BH    (KT_H*NT*32u*32u)                 // 1,572,864
#define OFF_BI   (OFF_BH + SZ_BH)
#define SZ_BI    (KT_E*NT*32u*32u)                 //   786,432
#define OFF_BIAS (OFF_BI + SZ_BI)
#define SZ_BIAS  (4u*H_DIM*4u)
#define OFF_XE   ((OFF_BIAS + SZ_BIAS + 255u) & ~255u)
#define SZ_XE    ((unsigned)T_DIM*4u*KT_E*32u*32u) // 33,554,432

#define WMMA_BF16(A,B,C) \
  __builtin_amdgcn_wmma_f32_16x16x32_bf16(false,(A),false,(B),(short)0,(C),false,false)

// ---------------------------------------------------------------------------
// Prep kernel 1: pack W_h* / W_i* into WMMA B-operand layout; fold biases.
// ---------------------------------------------------------------------------
__global__ void pack_weights(const float* __restrict__ Whr, const float* __restrict__ Whz,
                             const float* __restrict__ Whn, const float* __restrict__ Wir,
                             const float* __restrict__ Wiz, const float* __restrict__ Win,
                             const float* __restrict__ bir, const float* __restrict__ bhr,
                             const float* __restrict__ biz, const float* __restrict__ bhz,
                             const float* __restrict__ bin_, const float* __restrict__ bhn,
                             unsigned char* __restrict__ ws) {
    const int g = blockIdx.x * blockDim.x + threadIdx.x;
    __bf16* Bh   = (__bf16*)(ws + OFF_BH);
    __bf16* Bi   = (__bf16*)(ws + OFF_BI);
    float*  bias = (float*) (ws + OFF_BIAS);
    const int NBH = H_DIM * NG;   // 786432
    const int NBI = E_DIM * NG;   // 393216
    if (g < NBH) {
        const int k = g / NG, c = g - k * NG;
        const int gate = c >> 9, j = c & 511;
        const float* W = (gate == 0) ? Whr : (gate == 1) ? Whz : Whn;
        const float v = W[j * H_DIM + k];
        const int kt = k >> 5, nt = c >> 4, kk = k & 31, nn = c & 15;
        const int lane = nn + (((kk >> 3) & 1) << 4), rem = ((kk >> 4) << 3) + (kk & 7);
        Bh[((kt * NT + nt) * 32 + lane) * 16 + (rem >> 1) * 2 + (rem & 1)] = (__bf16)v;
    } else if (g < NBH + NBI) {
        const int g2 = g - NBH;
        const int k = g2 / NG, c = g2 - k * NG;
        const int gate = c >> 9, j = c & 511;
        const float* W = (gate == 0) ? Wir : (gate == 1) ? Wiz : Win;
        const float v = W[j * E_DIM + k];
        const int kt = k >> 5, nt = c >> 4, kk = k & 31, nn = c & 15;
        const int lane = nn + (((kk >> 3) & 1) << 4), rem = ((kk >> 4) << 3) + (kk & 7);
        Bi[((kt * NT + nt) * 32 + lane) * 16 + (rem >> 1) * 2 + (rem & 1)] = (__bf16)v;
    } else if (g < NBH + NBI + 4 * H_DIM) {
        const int g3 = g - (NBH + NBI);
        const int a = g3 >> 9, j = g3 & 511;
        float v;
        if      (a == 0) v = bir[j] + bhr[j];
        else if (a == 1) v = biz[j] + bhz[j];
        else if (a == 2) v = bin_[j];
        else             v = bhn[j];
        bias[a * H_DIM + j] = v;
    }
}

// ---------------------------------------------------------------------------
// Prep kernel 2: gather emb[x] and pack to bf16 WMMA A-operand layout.
// ---------------------------------------------------------------------------
__global__ void pack_xe(const int* __restrict__ x, const float* __restrict__ emb,
                        unsigned char* __restrict__ ws) {
    const int g = blockIdx.x * blockDim.x + threadIdx.x;   // g < B*T*E
    const int e = g & (E_DIM - 1);
    const int b = (g >> 8) & (B_DIM - 1);
    const int t = g >> 14;
    const int tok = x[b * T_DIM + t];
    const float v = emb[tok * E_DIM + e];
    const int m = b & 15, wg = b >> 4;
    const int kt = e >> 5, ko = e & 31;
    const int lane = m + (((ko >> 3) & 1) << 4);
    const int cc = ko >> 4, p = ko & 7;
    __bf16* xe = (__bf16*)(ws + OFF_XE);
    xe[(((t * 4 + wg) * KT_E + kt) * 32 + lane) * 16 + cc * 8 + p] = (__bf16)v;
}

// ---------------------------------------------------------------------------
// Persistent GRU scan. 4 WGs x 512 threads. WG wg owns batch rows
// [16*wg, 16*wg+16). Wave w owns h-columns [32*w, 32*w+32): it computes the
// r/z/n gate tiles for those columns and keeps h in D-layout registers.
// No inter-WG communication (batch rows are independent in a GRU).
// ---------------------------------------------------------------------------
__global__ void __launch_bounds__(512, 1)
gru_scan(const float* __restrict__ h0, const unsigned char* __restrict__ ws,
         const float* __restrict__ Wfc, const float* __restrict__ bfc,
         float* __restrict__ out) {
    const int tid  = threadIdx.x;
    const int w    = tid >> 5;        // wave 0..15
    const int lane = tid & 31;
    const int wg   = blockIdx.x;      // 0..3

    __shared__ __align__(32) __bf16 ldsA[KT_H * 512];   // 16 KB: h in A-layout
    __shared__ __align__(16) float  hbuf[16 * H_DIM];   // 32 KB: final head

    const float* bias = (const float*)(ws + OFF_BIAS);

    // Per-lane constants for the two 16-column blocks this wave owns.
    float br[2], bz[2], bn_i[2], bn_h[2];
    int   cg[2];
    float hold[2][8];                 // h in WMMA D layout (rows m, col = cg)
    const int mo = (lane >> 4) << 3;  // row offset from lane half

    #pragma unroll
    for (int cb = 0; cb < 2; ++cb) {
        const int c = (w * 2 + cb) * 16 + (lane & 15);
        cg[cb]   = c;
        br[cb]   = bias[c];
        bz[cb]   = bias[512 + c];
        bn_i[cb] = bias[1024 + c];
        bn_h[cb] = bias[1536 + c];
        #pragma unroll
        for (int i = 0; i < 8; ++i)
            hold[cb][i] = h0[(wg * 16 + (i + mo)) * H_DIM + c];
    }

    // Per-lane stream base offsets (bytes) for the two column blocks.
    const char* aBase = (const char*)ldsA + lane * 32;
    const unsigned laneB = (unsigned)lane * 32u;

    for (int t = 0; t < T_DIM; ++t) {
        // ---- phase A: scatter h (bf16) into WMMA A-operand layout in LDS ----
        #pragma unroll
        for (int cb = 0; cb < 2; ++cb) {
            const int c  = cg[cb];
            const int kt = c >> 5, ko = c & 31;
            const int hi = (ko >> 3) & 1, cc = ko >> 4, p = ko & 7;
            const int base = kt * 512 + (hi << 8) + (cc << 3) + p;
            #pragma unroll
            for (int i = 0; i < 8; ++i)
                ldsA[base + (i + mo) * 16] = (__bf16)hold[cb][i];
        }
        __syncthreads();

        const char* xeT = (const char*)ws + OFF_XE +
                          (unsigned)(t * 4 + wg) * (KT_E * 1024u) + laneB;

        // ---- phase B: gate GEMMs (WMMA) + in-register combine ----
        #pragma unroll
        for (int cb = 0; cb < 2; ++cb) {
            const int nth = w * 2 + cb;                  // h-column tile 0..31
            v8f accR  = {0.f,0.f,0.f,0.f,0.f,0.f,0.f,0.f};
            v8f accZ  = {0.f,0.f,0.f,0.f,0.f,0.f,0.f,0.f};
            v8f accNx = {0.f,0.f,0.f,0.f,0.f,0.f,0.f,0.f};
            v8f accNh = {0.f,0.f,0.f,0.f,0.f,0.f,0.f,0.f};

            // One running pointer per stream; R/Z/N at fixed byte distances.
            const char* pB = (const char*)ws + OFF_BH + (unsigned)nth * 1024u + laneB;
            int aoff = 0;
            #pragma unroll 2
            for (int kt = 0; kt < KT_H; ++kt) {          // hidden projections
                const v16bf a  = *(const v16bf*)(aBase + aoff);
                const v16bf bR = *(const v16bf*)(pB);
                const v16bf bZ = *(const v16bf*)(pB + ZOFF);
                const v16bf bN = *(const v16bf*)(pB + NOFF);
                __builtin_prefetch(pB + 2 * KSTRIDE, 0, 1);
                __builtin_prefetch(pB + 2 * KSTRIDE + ZOFF, 0, 1);
                __builtin_prefetch(pB + 2 * KSTRIDE + NOFF, 0, 1);
                accR  = WMMA_BF16(a, bR, accR);
                accZ  = WMMA_BF16(a, bZ, accZ);
                accNh = WMMA_BF16(a, bN, accNh);
                pB   += KSTRIDE;
                aoff += 1024;
            }
            const char* pBi = (const char*)ws + OFF_BI + (unsigned)nth * 1024u + laneB;
            const char* pX  = xeT;
            #pragma unroll 2
            for (int kt = 0; kt < KT_E; ++kt) {          // input projections
                const v16bf a  = *(const v16bf*)(pX);
                const v16bf bR = *(const v16bf*)(pBi);
                const v16bf bZ = *(const v16bf*)(pBi + ZOFF);
                const v16bf bN = *(const v16bf*)(pBi + NOFF);
                accR  = WMMA_BF16(a, bR, accR);
                accZ  = WMMA_BF16(a, bZ, accZ);
                accNx = WMMA_BF16(a, bN, accNx);
                pBi  += KSTRIDE;
                pX   += 1024;
            }
            #pragma unroll
            for (int i = 0; i < 8; ++i) {
                const float r  = 1.f / (1.f + __expf(-(accR[i] + br[cb])));
                const float z  = 1.f / (1.f + __expf(-(accZ[i] + bz[cb])));
                const float np = accNx[i] + bn_i[cb] + r * (accNh[i] + bn_h[cb]);
                const float ex = __expf(-2.f * np);
                const float n  = (1.f - ex) / (1.f + ex);     // tanh(np)
                hold[cb][i] = n + z * (hold[cb][i] - n);
            }
        }
        __syncthreads();
    }

    // ---- epilogue: write hT and the sigmoid head ----
    #pragma unroll
    for (int cb = 0; cb < 2; ++cb) {
        const int c = cg[cb];
        #pragma unroll
        for (int i = 0; i < 8; ++i) {
            const int m = i + mo;
            hbuf[m * H_DIM + c] = hold[cb][i];
            out[64 + (wg * 16 + m) * H_DIM + c] = hold[cb][i];
        }
    }
    __syncthreads();
    if (tid < 16) {
        float acc = bfc[0];
        for (int j = 0; j < H_DIM; ++j)
            acc += hbuf[tid * H_DIM + j] * Wfc[j];
        out[wg * 16 + tid] = 1.f / (1.f + __expf(-acc));
    }
}

// ---------------------------------------------------------------------------
extern "C" void kernel_launch(void* const* d_in, const int* in_sizes, int n_in,
                              void* d_out, int out_size, void* d_ws, size_t ws_size,
                              hipStream_t stream) {
    const int*   x    = (const int*)  d_in[0];
    const float* h0   = (const float*)d_in[1];
    const float* emb  = (const float*)d_in[2];
    const float* Wir  = (const float*)d_in[3],  *bir = (const float*)d_in[4];
    const float* Whr  = (const float*)d_in[5],  *bhr = (const float*)d_in[6];
    const float* Wiz  = (const float*)d_in[7],  *biz = (const float*)d_in[8];
    const float* Whz  = (const float*)d_in[9],  *bhz = (const float*)d_in[10];
    const float* Win  = (const float*)d_in[11], *bin_ = (const float*)d_in[12];
    const float* Whn  = (const float*)d_in[13], *bhn = (const float*)d_in[14];
    const float* Wfc  = (const float*)d_in[15], *bfc = (const float*)d_in[16];
    unsigned char* ws = (unsigned char*)d_ws;
    float* out = (float*)d_out;

    const int n1 = H_DIM * NG + E_DIM * NG + 4 * H_DIM;
    pack_weights<<<(n1 + 255) / 256, 256, 0, stream>>>(
        Whr, Whz, Whn, Wir, Wiz, Win, bir, bhr, biz, bhz, bin_, bhn, ws);

    const int n2 = B_DIM * T_DIM * E_DIM;                 // 16,777,216
    pack_xe<<<n2 / 256, 256, 0, stream>>>(x, emb, ws);

    gru_scan<<<4, 512, 0, stream>>>(h0, ws, Wfc, bfc, out);
}
